// GCNEncoder_81905026334999
// MI455X (gfx1250) — compile-verified, matrix operand order checked by reading, not run on hardware
//
#include <hip/hip_runtime.h>
#include <hip/hip_bf16.h>
#include <stdint.h>

#define N_NODES 50000
#define N_EDGES 1600000
#define IN_DIM  512
#define D1      256
#define HID     128

typedef __attribute__((ext_vector_type(16))) __bf16 v16bf;
typedef __attribute__((ext_vector_type(8)))  __bf16 v8bf;
typedef __attribute__((ext_vector_type(8)))  float  v8f;

__device__ __forceinline__ void atomAddF(float* p, float v) {
  __hip_atomic_fetch_add(p, v, __ATOMIC_RELAXED, __HIP_MEMORY_SCOPE_AGENT);
}

// ---------------- degree / dis ----------------
__global__ void k_degree(const long long* __restrict__ dst, float* __restrict__ deg, int E) {
  int i = blockIdx.x * blockDim.x + threadIdx.x;
  if (i < E) atomAddF(&deg[dst[i]], 1.0f);
}

__global__ void k_dis(float* __restrict__ deg, int n) {
  int i = blockIdx.x * blockDim.x + threadIdx.x;
  if (i < n) deg[i] = rsqrtf(deg[i] + 1.0f);   // +1 = self loop; always > 0
}

// -------- weight prep: W[K,N] f32 -> Wt_hi/lo[N,K] bf16 split --------
__global__ void k_prepw(const float* __restrict__ W, __bf16* __restrict__ hi,
                        __bf16* __restrict__ lo, int K, int N) {
  int i = blockIdx.x * blockDim.x + threadIdx.x;
  if (i >= K * N) return;
  int k = i % K, n = i / K;
  float w = W[(size_t)k * N + n];
  __bf16 h = (__bf16)w;
  hi[i] = h;
  lo[i] = (__bf16)(w - (float)h);
}

// A-fragment (16-bit A 16x32 layout): this lane covers K runs [kb, kb+8) and
// [kb+16, kb+24). Load fp32, split into bf16 hi + lo.
__device__ __forceinline__ void load_split_a(const float* __restrict__ p, v16bf& hi, v16bf& lo) {
  float af[16];
  const float4 a0 = *(const float4*)(p);
  const float4 a1 = *(const float4*)(p + 4);
  const float4 a2 = *(const float4*)(p + 16);
  const float4 a3 = *(const float4*)(p + 20);
  af[0]=a0.x;  af[1]=a0.y;  af[2]=a0.z;  af[3]=a0.w;
  af[4]=a1.x;  af[5]=a1.y;  af[6]=a1.z;  af[7]=a1.w;
  af[8]=a2.x;  af[9]=a2.y;  af[10]=a2.z; af[11]=a2.w;
  af[12]=a3.x; af[13]=a3.y; af[14]=a3.z; af[15]=a3.w;
  #pragma unroll
  for (int i = 0; i < 16; ++i) {
    __bf16 h = (__bf16)af[i];
    hi[i] = h;
    lo[i] = (__bf16)(af[i] - (float)h);
  }
}

// ---------------- split-bf16 WMMA GEMM ----------------
// Wave = 2 M-tiles x NT N-tiles: two independent accumulator chains hide the
// WMMA->WMMA RAW hazard, and each B fragment is reused by 2 WMMA triplets.
template<int K, int N, int NT>
__global__ __launch_bounds__(256) void k_gemm(const float* __restrict__ X,
                                              const __bf16* __restrict__ Whi,
                                              const __bf16* __restrict__ Wlo,
                                              float* __restrict__ H, int Mtiles) {
  const int wave = threadIdx.x >> 5;
  const int lane = threadIdx.x & 31;
  const int pair = blockIdx.x * 8 + wave;
  const int m0 = pair * 2;
  if (m0 >= Mtiles) return;                   // wave-uniform: EXEC stays all-ones
  const bool has1 = (m0 + 1 < Mtiles);
  const int m1v = has1 ? (m0 + 1) : m0;       // clamp loads of the dead tile
  const int half = lane >> 4;                 // 0: K runs 0-7/16-23, 1: 8-15/24-31
  const int l16  = lane & 15;
  const int ncb  = blockIdx.y * (NT * 16);
  const float* xr0 = X + (size_t)(m0  * 16 + l16) * K;
  const float* xr1 = X + (size_t)(m1v * 16 + l16) * K;

  v8f acc0[NT] = {}, acc1[NT] = {};
  for (int kk = 0; kk < K; kk += 32) {
    const int kb = kk + half * 8;
    v16bf a0h, a0l, a1h, a1l;
    load_split_a(xr0 + kb, a0h, a0l);
    load_split_a(xr1 + kb, a1h, a1l);
    #pragma unroll
    for (int t = 0; t < NT; ++t) {
      // B fragment (16-bit B 32x16 layout mirrors A): two 16B runs per lane
      const size_t wb = (size_t)(ncb + t * 16 + l16) * K + kb;
      v8bf bh0 = *(const v8bf*)(Whi + wb);
      v8bf bh1 = *(const v8bf*)(Whi + wb + 16);
      v8bf bl0 = *(const v8bf*)(Wlo + wb);
      v8bf bl1 = *(const v8bf*)(Wlo + wb + 16);
      v16bf bhi = __builtin_shufflevector(bh0, bh1, 0,1,2,3,4,5,6,7,8,9,10,11,12,13,14,15);
      v16bf blo = __builtin_shufflevector(bl0, bl1, 0,1,2,3,4,5,6,7,8,9,10,11,12,13,14,15);
      acc0[t] = __builtin_amdgcn_wmma_f32_16x16x32_bf16(false, a0h, false, bhi, (short)0, acc0[t], false, false);
      acc1[t] = __builtin_amdgcn_wmma_f32_16x16x32_bf16(false, a1h, false, bhi, (short)0, acc1[t], false, false);
      acc0[t] = __builtin_amdgcn_wmma_f32_16x16x32_bf16(false, a0h, false, blo, (short)0, acc0[t], false, false);
      acc1[t] = __builtin_amdgcn_wmma_f32_16x16x32_bf16(false, a1h, false, blo, (short)0, acc1[t], false, false);
      acc0[t] = __builtin_amdgcn_wmma_f32_16x16x32_bf16(false, a0l, false, bhi, (short)0, acc0[t], false, false);
      acc1[t] = __builtin_amdgcn_wmma_f32_16x16x32_bf16(false, a1l, false, bhi, (short)0, acc1[t], false, false);
    }
  }
  // C/D layout: VGPR r -> row r (lanes 0-15) / row r+8 (lanes 16-31), col = lane%16
  {
    const int mrow0 = m0 * 16 + half * 8;
    #pragma unroll
    for (int t = 0; t < NT; ++t) {
      const int col = ncb + t * 16 + l16;
      #pragma unroll
      for (int r = 0; r < 8; ++r)
        H[(size_t)(mrow0 + r) * N + col] = acc0[t][r];
    }
  }
  if (has1) {
    const int mrow1 = (m0 + 1) * 16 + half * 8;
    #pragma unroll
    for (int t = 0; t < NT; ++t) {
      const int col = ncb + t * 16 + l16;
      #pragma unroll
      for (int r = 0; r < 8; ++r)
        H[(size_t)(mrow1 + r) * N + col] = acc1[t][r];
    }
  }
}

// ---------------- edge aggregation (wave-per-edge, L2-resident H) ----------------
template<int C>
__global__ void k_edge_agg(const long long* __restrict__ src, const long long* __restrict__ dst,
                           const float* __restrict__ dis, const float* __restrict__ H,
                           float* __restrict__ agg, int E) {
  const int lane = threadIdx.x & 31;
  int wid = (blockIdx.x * blockDim.x + threadIdx.x) >> 5;
  const int nw = (gridDim.x * blockDim.x) >> 5;
  for (int e = wid; e < E; e += nw) {
    const long long s = src[e], d = dst[e];           // wave-uniform
    const float coeff = dis[s] * dis[d];
    const float* hs = H + (size_t)s * C;
    float* od = agg + (size_t)d * C;
    #pragma unroll
    for (int j = 0; j < C / 128; ++j) {
      const int c = j * 128 + lane * 4;
      float4 v = *(const float4*)(hs + c);
      atomAddF(od + c + 0, coeff * v.x);
      atomAddF(od + c + 1, coeff * v.y);
      atomAddF(od + c + 2, coeff * v.z);
      atomAddF(od + c + 3, coeff * v.w);
    }
  }
}

// ---------------- self-loop + bias (no atomics needed) ----------------
template<int C>
__global__ void k_self_bias(float* __restrict__ agg, const float* __restrict__ H,
                            const float* __restrict__ dis, const float* __restrict__ b, int n) {
  long long i = (long long)blockIdx.x * blockDim.x + threadIdx.x;
  if (i >= (long long)n * C) return;
  int node = (int)(i / C), c = (int)(i % C);
  float w = dis[node];
  agg[i] += w * w * H[i] + b[c];
}

// ---------------- per-column sum / sumsq ----------------
template<int C>
__global__ void k_stats(const float* __restrict__ H, float* __restrict__ sum,
                        float* __restrict__ sq, int n) {
  const int c = threadIdx.x;                 // blockDim == C
  float s = 0.f, s2 = 0.f;
  for (int r = blockIdx.x; r < n; r += gridDim.x) {
    float v = H[(size_t)r * C + c];
    s += v; s2 += v * v;
  }
  atomAddF(&sum[c], s);
  atomAddF(&sq[c], s2);
}

// ---------------- batchnorm (+optional relu) ----------------
template<int C, bool RELU>
__global__ void k_bn(const float* __restrict__ in, float* __restrict__ out,
                     const float* __restrict__ sum, const float* __restrict__ sq,
                     const float* __restrict__ gamma, const float* __restrict__ beta, int n) {
  long long i = (long long)blockIdx.x * blockDim.x + threadIdx.x;
  if (i >= (long long)n * C) return;
  int c = (int)(i % C);
  const float invN = 1.0f / (float)n;
  float mean = sum[c] * invN;
  float var  = sq[c] * invN - mean * mean;   // biased variance, matches reference
  float y = (in[i] - mean) * rsqrtf(var + 1e-5f) * gamma[c] + beta[c];
  if (RELU) y = fmaxf(y, 0.0f);
  out[i] = y;
}

extern "C" void kernel_launch(void* const* d_in, const int* in_sizes, int n_in,
                              void* d_out, int out_size, void* d_ws, size_t ws_size,
                              hipStream_t stream) {
  (void)in_sizes; (void)n_in; (void)out_size; (void)ws_size;
  const float*     x   = (const float*)d_in[0];
  const long long* ei  = (const long long*)d_in[1];   // int64 per reference
  const float*     W1  = (const float*)d_in[2];
  const float*     b1  = (const float*)d_in[3];
  const float*     g1  = (const float*)d_in[4];
  const float*     be1 = (const float*)d_in[5];
  const float*     W2  = (const float*)d_in[6];
  const float*     b2  = (const float*)d_in[7];
  const float*     g2  = (const float*)d_in[8];
  const float*     be2 = (const float*)d_in[9];
  const long long* src = ei;
  const long long* dst = ei + N_EDGES;

  char* ws = (char*)d_ws;
  size_t off = 0;
  auto take = [&](size_t bytes) -> char* {
    char* p = ws + off;
    off = (off + bytes + 255) & ~(size_t)255;
    return p;
  };
  float*  deg   = (float*)take((size_t)N_NODES * 4);          // becomes dis in-place
  __bf16* w1hi  = (__bf16*)take((size_t)IN_DIM * D1 * 2);
  __bf16* w1lo  = (__bf16*)take((size_t)IN_DIM * D1 * 2);
  __bf16* w2hi  = (__bf16*)take((size_t)D1 * HID * 2);
  __bf16* w2lo  = (__bf16*)take((size_t)D1 * HID * 2);
  float*  stats = (float*)take(4 * 256 * 4);
  float*  sum1 = stats, *sq1 = stats + 256, *sum2 = stats + 512, *sq2 = stats + 640;
  float*  aggA = (float*)take((size_t)N_NODES * D1 * 4);      // 51.2MB: agg1 -> normalized H1n
  float*  bufH = (float*)take((size_t)N_NODES * D1 * 4);      // 51.2MB: H1, later H2 + agg2
  float*  H1   = bufH;
  float*  H2   = bufH;                                        // overlays dead H1
  float*  agg2 = bufH + (size_t)N_NODES * HID;                // second half of bufH

  // init
  hipMemsetAsync(deg,   0, (size_t)N_NODES * 4, stream);
  hipMemsetAsync(stats, 0, 4 * 256 * 4, stream);
  hipMemsetAsync(aggA,  0, (size_t)N_NODES * D1 * 4, stream);

  // graph normalization terms
  k_degree<<<(N_EDGES + 255) / 256, 256, 0, stream>>>(dst, deg, N_EDGES);
  k_dis<<<(N_NODES + 255) / 256, 256, 0, stream>>>(deg, N_NODES);

  // weight prep (split-bf16, transposed to [N][K])
  k_prepw<<<(IN_DIM * D1 + 255) / 256, 256, 0, stream>>>(W1, w1hi, w1lo, IN_DIM, D1);
  k_prepw<<<(D1 * HID + 255) / 256, 256, 0, stream>>>(W2, w2hi, w2lo, D1, HID);

  const int MT    = N_NODES / 16;        // 3125, exact
  const int PAIRS = (MT + 1) / 2;        // 1563
  const int GX    = (PAIRS + 7) / 8;     // 8 waves/block, 2 M-tiles/wave

  // ---- layer 1 ----  (NT=4 -> 64 cols per wave; grid.y = 256/64 = 4)
  k_gemm<IN_DIM, D1, 4><<<dim3(GX, 4), 256, 0, stream>>>(x, w1hi, w1lo, H1, MT);
  k_edge_agg<D1><<<4096, 256, 0, stream>>>(src, dst, deg, H1, aggA, N_EDGES);
  k_self_bias<D1><<<(int)(((long long)N_NODES * D1 + 255) / 256), 256, 0, stream>>>(aggA, H1, deg, b1, N_NODES);
  k_stats<D1><<<512, D1, 0, stream>>>(aggA, sum1, sq1, N_NODES);
  k_bn<D1, true><<<(int)(((long long)N_NODES * D1 + 255) / 256), 256, 0, stream>>>(aggA, aggA, sum1, sq1, g1, be1, N_NODES);

  // ---- layer 2 ----  (grid.y = 128/64 = 2)
  k_gemm<D1, HID, 4><<<dim3(GX, 2), 256, 0, stream>>>(aggA, w2hi, w2lo, H2, MT);
  hipMemsetAsync(agg2, 0, (size_t)N_NODES * HID * 4, stream);
  k_edge_agg<HID><<<4096, 256, 0, stream>>>(src, dst, deg, H2, agg2, N_EDGES);
  k_self_bias<HID><<<(int)(((long long)N_NODES * HID + 255) / 256), 256, 0, stream>>>(agg2, H2, deg, b2, N_NODES);
  k_stats<HID><<<512, HID, 0, stream>>>(agg2, sum2, sq2, N_NODES);
  k_bn<HID, false><<<(int)(((long long)N_NODES * HID + 255) / 256), 256, 0, stream>>>(agg2, (float*)d_out, sum2, sq2, g2, be2, N_NODES);
}
